// GraphEncoder_20169166422561
// MI455X (gfx1250) — matrix operand order repaired and checked by reading, not compiled, not run on hardware
//
#include <hip/hip_runtime.h>
#include <hip/hip_bf16.h>

typedef __attribute__((ext_vector_type(16))) _Float16     v16h;
typedef __attribute__((ext_vector_type(8)))  float        v8f;
typedef __attribute__((ext_vector_type(4)))  unsigned int v4u;
typedef __attribute__((ext_vector_type(8)))  unsigned int v8u;

#define U_  50000
#define NI_ 100000
#define NA_ 20000
#define EMB 64
#define HID 100
#define B_  32
#define N_  256
#define R_  10
#define S_  257   // N_ + 1

__device__ __forceinline__ float lrelu(float x) { return x > 0.0f ? x : 0.2f * x; }

// ---------------------------------------------------------------------------
// Degree counting: deg[idx[e]] += 1  (hardware global_atomic_add_f32)
// ---------------------------------------------------------------------------
__global__ void count_deg_kernel(const int* __restrict__ idx, int n, float* __restrict__ deg) {
    int i = blockIdx.x * blockDim.x + threadIdx.x;
    if (i < n) unsafeAtomicAdd(&deg[idx[i]], 1.0f);
}

// ---------------------------------------------------------------------------
// Edge aggregation: agg[dst] += X[src] * rsqrt(max(deg_out[src],1))
// one thread per (edge, 4 floats): b128 gather + 4 f32 hardware atomics
// ---------------------------------------------------------------------------
__global__ void edge_aggregate_kernel(const int* __restrict__ src, const int* __restrict__ dst,
                                      int nE, const float* __restrict__ X,
                                      const float* __restrict__ deg_out, float* __restrict__ agg) {
    long long tid = (long long)blockIdx.x * blockDim.x + threadIdx.x;
    if (tid >= (long long)nE * (EMB / 4)) return;
    int e = (int)(tid >> 4);
    int q = ((int)tid & 15) << 2;
    int s = src[e], d = dst[e];
    float sc = rsqrtf(fmaxf(deg_out[s], 1.0f));
    const float4 x = *(const float4*)(X + (size_t)s * EMB + q);
    float* ap = agg + (size_t)d * EMB + q;
    unsafeAtomicAdd(ap + 0, x.x * sc);
    unsafeAtomicAdd(ap + 1, x.y * sc);
    unsafeAtomicAdd(ap + 2, x.z * sc);
    unsafeAtomicAdd(ap + 3, x.w * sc);
}

// ---------------------------------------------------------------------------
// Row GEMM via WMMA f32_16x16x32_f16:  P = (scale(X[row])) @ W(64x100) (+bias)
// MODE 0: out = lrelu(P)        MODE 1: out = 0.5*P
// MODE 2: out = lrelu(out+0.5P) MODE 3: out = P
// Two independent accumulators break the WMMA->WMMA RAW hazard chain.
// ---------------------------------------------------------------------------
template <int MODE, bool IDX, bool DEG, bool BIAS>
__global__ void gemm_rows_wmma(const float* __restrict__ X, const int* __restrict__ rowidx,
                               const float* __restrict__ deg, const float* __restrict__ W,
                               const float* __restrict__ bias, float* __restrict__ out,
                               int M) {
    int warp = threadIdx.x >> 5;
    int lane = threadIdx.x & 31;
    int mtile = blockIdx.x * (blockDim.x >> 5) + warp;
    if (mtile * 16 >= M) return;                      // wave-uniform exit
    int ntile = blockIdx.y;

    int mrow = mtile * 16 + (lane & 15);
    int half = lane >> 4;
    int srow = IDX ? rowidx[mrow] : mrow;
    const float* xr = X + (size_t)srow * EMB;
    float sc = DEG ? rsqrtf(fmaxf(deg[mrow], 1.0f)) : 1.0f;

    v16h a0, a1;
    #pragma unroll
    for (int v = 0; v < 8; ++v) {
        int k = ((v < 4) ? 0 : 16) + half * 8 + 2 * (v & 3);
        float2 x0 = *(const float2*)(xr + k);
        float2 x1 = *(const float2*)(xr + k + 32);
        a0[2 * v]     = (_Float16)(x0.x * sc);
        a0[2 * v + 1] = (_Float16)(x0.y * sc);
        a1[2 * v]     = (_Float16)(x1.x * sc);
        a1[2 * v + 1] = (_Float16)(x1.y * sc);
    }

    int col = ntile * 16 + (lane & 15);
    bool cv = col < HID;
    int colc = cv ? col : (HID - 1);                  // clamp: load in-bounds, select after
    v16h b0, b1;
    #pragma unroll
    for (int v = 0; v < 8; ++v) {
        int kr = ((v < 4) ? 0 : 16) + half * 8 + 2 * (v & 3);
        float w00 = W[(size_t)kr * HID + colc];
        float w01 = W[(size_t)(kr + 1) * HID + colc];
        float w10 = W[(size_t)(kr + 32) * HID + colc];
        float w11 = W[(size_t)(kr + 33) * HID + colc];
        b0[2 * v]     = (_Float16)(cv ? w00 : 0.0f);
        b0[2 * v + 1] = (_Float16)(cv ? w01 : 0.0f);
        b1[2 * v]     = (_Float16)(cv ? w10 : 0.0f);
        b1[2 * v + 1] = (_Float16)(cv ? w11 : 0.0f);
    }

    v8f c0 = {}, c1 = {};
    c0 = __builtin_amdgcn_wmma_f32_16x16x32_f16(false, a0, false, b0, (short)0, c0, false, false);
    c1 = __builtin_amdgcn_wmma_f32_16x16x32_f16(false, a1, false, b1, (short)0, c1, false, false);

    float bval = BIAS ? bias[colc] : 0.0f;
    if (cv) {
        #pragma unroll
        for (int r = 0; r < 8; ++r) {
            int m = mtile * 16 + half * 8 + r;
            float val = c0[r] + c1[r] + bval;
            size_t o = (size_t)m * HID + col;
            if (MODE == 0)      out[o] = lrelu(val);
            else if (MODE == 1) out[o] = 0.5f * val;
            else if (MODE == 2) out[o] = lrelu(out[o] + 0.5f * val);
            else                out[o] = val;
        }
    }
}

// ---------------------------------------------------------------------------
// Batched WMMA GEMM: out[b] = adj[b](256x256) @ support[b](256x100) + gcn_b
// One block per (batch, n-tile). The 256x16 f32 support panel is DMA'd into
// LDS by the Tensor Data Mover (tensor_load_to_lds, D# per ISA 08 sec.8),
// converted once to f16, then reused by 4 waves sweeping all 16 m-tiles.
// OOB columns (n-tile 6 covers cols 96..111, tensor_dim0=4) read as zero by TDM.
// fused: seq = (1-eps)*out + eps*gnn[cur_node] -> interest[b, 0..255, :]
// ---------------------------------------------------------------------------
__global__ void gemm_adj_wmma(const float* __restrict__ adj, const float* __restrict__ support,
                              const float* __restrict__ gcn_b, const float* __restrict__ eps_p,
                              const float* __restrict__ gnn, const int* __restrict__ cur_node,
                              float* __restrict__ interest) {
    // single static LDS block => f32 panel sits at LDS offset 0 (D#.lds_addr = 0)
    __shared__ float smem[N_ * 16 + N_ * 8];          // 16KB f32 panel + 8KB f16 panel
    float*     sBf = smem;                            // [k][n] f32, offset 0
    _Float16*  sB  = (_Float16*)(smem + N_ * 16);     // [k][n] f16

    int bb = blockIdx.x / 7;
    int nt = blockIdx.x % 7;
    int t = threadIdx.x;
    int warp = t >> 5;
    int lane = t & 31;

    const float* A = adj + (size_t)bb * N_ * N_;
    const float* Bp = support + (size_t)bb * N_ * HID;

    // ---- TDM: async 2D tile load (16 cols x 256 rows, f32) into LDS offset 0
    if (t < 32) {                                     // wave 0 issues; EXEC ignored by TDM
        unsigned long long ga = (unsigned long long)(Bp + (size_t)nt * 16);
        unsigned int tdim0 = (unsigned int)(HID - nt * 16);   // valid cols from tile start
        v4u g0;
        g0[0] = 1u;                                           // count=1 (valid user D#)
        g0[1] = 0u;                                           // lds_addr = 0
        g0[2] = (unsigned int)(ga & 0xFFFFFFFFu);             // global_addr[31:0]
        g0[3] = (unsigned int)((ga >> 32) & 0x01FFFFFFu) | (2u << 30);  // addr[56:32], type=2
        v8u g1;
        g1[0] = (2u << 16);                                   // data_size=2 (4 bytes)
        g1[1] = (tdim0 & 0xFFFFu) << 16;                      // tensor_dim0[15:0]
        g1[2] = ((unsigned int)N_ << 16) | (tdim0 >> 16);     // tensor_dim1=256 | dim0[31:16]
        g1[3] = (16u << 16);                                  // tile_dim0=16
        g1[4] = (unsigned int)N_;                             // tile_dim1=256
        g1[5] = (unsigned int)HID;                            // tensor_dim0_stride=100
        g1[6] = 0u;
        g1[7] = 0u;
        asm volatile("tensor_load_to_lds %0, %1" :: "s"(g0), "s"(g1) : "memory");
        __builtin_amdgcn_s_wait_tensorcnt(0);
    }
    __syncthreads();

    // ---- convert f32 panel -> f16 panel (once per block)
    #pragma unroll
    for (int i = 0; i < (N_ * 16) / 128; ++i) {
        int linear = i * 128 + t;
        sB[linear] = (_Float16)sBf[linear];
    }
    __syncthreads();

    int half = lane >> 4;
    int col = nt * 16 + (lane & 15);
    bool cv = col < HID;
    float e = eps_p[0];
    float gb = gcn_b[cv ? col : (HID - 1)];

    for (int mt = warp; mt < 16; mt += 4) {
        int mrow = mt * 16 + (lane & 15);
        v8f c0 = {}, c1 = {};                         // dual accumulators: no RAW chain
        #pragma unroll
        for (int ks = 0; ks < 8; ++ks) {
            int kb = ks * 32;
            v16h a, b;
            #pragma unroll
            for (int v = 0; v < 8; ++v) {
                int k = kb + ((v < 4) ? 0 : 16) + half * 8 + 2 * (v & 3);
                float2 av = *(const float2*)(A + (size_t)mrow * N_ + k);
                a[2 * v]     = (_Float16)av.x;
                a[2 * v + 1] = (_Float16)av.y;
                int kr = kb + ((v < 4) ? 0 : 16) + half * 8 + 2 * (v & 3);
                b[2 * v]     = sB[kr * 16 + (lane & 15)];
                b[2 * v + 1] = sB[(kr + 1) * 16 + (lane & 15)];
            }
            if (ks & 1)
                c1 = __builtin_amdgcn_wmma_f32_16x16x32_f16(false, a, false, b, (short)0, c1, false, false);
            else
                c0 = __builtin_amdgcn_wmma_f32_16x16x32_f16(false, a, false, b, (short)0, c0, false, false);
        }
        if (cv) {
            #pragma unroll
            for (int r = 0; r < 8; ++r) {
                int m = mt * 16 + half * 8 + r;
                float val = c0[r] + c1[r] + gb;
                int cn = cur_node[bb * N_ + m];
                float seq = (1.0f - e) * val + e * gnn[(size_t)cn * HID + col];
                interest[((size_t)bb * S_ + m) * HID + col] = seq;
            }
        }
    }
}

// ---------------------------------------------------------------------------
// Rejection row: mm = relu(((mean gnn[rej_f] + mean gnn[rej_i])) @ fcW + fcb)
// writes interest[b, 256, :]
// ---------------------------------------------------------------------------
__global__ void rej_mm_kernel(const float* __restrict__ gnn, const int* __restrict__ rej_f,
                              const int* __restrict__ rej_i, const float* __restrict__ fcW,
                              const float* __restrict__ fcb, float* __restrict__ interest) {
    int b = blockIdx.x;
    int t = threadIdx.x;
    __shared__ float v[HID];
    if (t < HID) {
        float acc = 0.0f;
        for (int r = 0; r < R_; ++r) acc += gnn[(size_t)rej_f[b * R_ + r] * HID + t];
        for (int r = 0; r < R_; ++r) acc += gnn[(size_t)rej_i[b * R_ + r] * HID + t];
        v[t] = acc * (1.0f / R_);
    }
    __syncthreads();
    if (t < HID) {
        float o = fcb[t];
        for (int k = 0; k < HID; ++k) o += v[k] * fcW[(size_t)k * HID + t];
        interest[((size_t)b * S_ + N_) * HID + t] = fmaxf(o, 0.0f);
    }
}

// ---------------------------------------------------------------------------
// Two-pass additive attention, one block per (k, b).
// ---------------------------------------------------------------------------
__global__ void attention_kernel(const float* __restrict__ interest,
                                 const float* __restrict__ Wi, const float* __restrict__ bi,
                                 const float* __restrict__ Wa, const float* __restrict__ ba,
                                 float* __restrict__ out) {
    int k = blockIdx.x / B_;
    int b = blockIdx.x % B_;
    int t = threadIdx.x;
    const float* I = interest + (size_t)b * S_ * HID;

    __shared__ float w[S_];
    __shared__ float red[512];
    __shared__ float c0[HID];

    // ---- pass 0 scores
    float d = -1e30f;
    if (t < S_) {
        d = bi[k];
        for (int h = 0; h < HID; ++h) d += I[(size_t)t * HID + h] * Wi[k * HID + h];
    }
    red[t] = d; __syncthreads();
    for (int s = 256; s > 0; s >>= 1) { if (t < s) red[t] = fmaxf(red[t], red[t + s]); __syncthreads(); }
    float mx = red[0]; __syncthreads();
    float e = (t < S_) ? expf(d - mx) : 0.0f;
    red[t] = e; __syncthreads();
    for (int s = 256; s > 0; s >>= 1) { if (t < s) red[t] += red[t + s]; __syncthreads(); }
    float sm = red[0]; __syncthreads();
    if (t < S_) w[t] = e / sm;
    __syncthreads();

    // ---- ctx0
    if (t < HID) {
        float acc = 0.0f;
        for (int s = 0; s < S_; ++s) acc += w[s] * I[(size_t)s * HID + t];
        c0[t] = acc;
    }
    __syncthreads();

    // ---- dot(ctx0, Wa_a[k])
    float pa = (t < HID) ? c0[t] * Wa[k * 2 * HID + HID + t] : 0.0f;
    red[t] = pa; __syncthreads();
    for (int s = 256; s > 0; s >>= 1) { if (t < s) red[t] += red[t + s]; __syncthreads(); }
    float dota = red[0]; __syncthreads();

    // ---- pass 1 scores: x@Wa_x + dota + ba + w0
    float d1 = -1e30f;
    if (t < S_) {
        d1 = ba[k] + dota + w[t];
        for (int h = 0; h < HID; ++h) d1 += I[(size_t)t * HID + h] * Wa[k * 2 * HID + h];
    }
    red[t] = d1; __syncthreads();
    for (int s = 256; s > 0; s >>= 1) { if (t < s) red[t] = fmaxf(red[t], red[t + s]); __syncthreads(); }
    float mx1 = red[0]; __syncthreads();
    float e1 = (t < S_) ? expf(d1 - mx1) : 0.0f;
    red[t] = e1; __syncthreads();
    for (int s = 256; s > 0; s >>= 1) { if (t < s) red[t] += red[t + s]; __syncthreads(); }
    float sm1 = red[0]; __syncthreads();
    if (t < S_) w[t] = e1 / sm1;
    __syncthreads();

    // ---- ctx1 -> out[k, b, 0, :]
    if (t < HID) {
        float acc = 0.0f;
        for (int s = 0; s < S_; ++s) acc += w[s] * I[(size_t)s * HID + t];
        out[(size_t)k * B_ * HID + (size_t)b * HID + t] = acc;
    }
}

// ---------------------------------------------------------------------------
extern "C" void kernel_launch(void* const* d_in, const int* in_sizes, int n_in,
                              void* d_out, int out_size, void* d_ws, size_t ws_size,
                              hipStream_t stream) {
    const float* emb   = (const float*)d_in[0];
    const float* W_int = (const float*)d_in[1];  const float* b_int = (const float*)d_in[2];
    const float* W_frd = (const float*)d_in[3];  const float* b_frd = (const float*)d_in[4];
    const float* W_lik = (const float*)d_in[5];  const float* b_lik = (const float*)d_in[6];
    const float* W_bel = (const float*)d_in[7];  const float* b_bel = (const float*)d_in[8];
    const float* gcn_W = (const float*)d_in[9];  const float* gcn_b = (const float*)d_in[10];
    const float* eps_p = (const float*)d_in[11];
    const float* fcW   = (const float*)d_in[12]; const float* fcb   = (const float*)d_in[13];
    const float* aiW   = (const float*)d_in[14]; const float* aib   = (const float*)d_in[15];
    const float* aaW   = (const float*)d_in[16]; const float* aab   = (const float*)d_in[17];
    const float* adj   = (const float*)d_in[18];
    const int* int_src = (const int*)d_in[19];   const int* int_dst = (const int*)d_in[20];
    const int* frd_src = (const int*)d_in[21];   const int* frd_dst = (const int*)d_in[22];
    const int* lik_src = (const int*)d_in[23];   const int* lik_dst = (const int*)d_in[24];
    const int* bel_src = (const int*)d_in[25];   const int* bel_dst = (const int*)d_in[26];
    const int* neighbors = (const int*)d_in[27]; const int* cur_node = (const int*)d_in[28];
    const int* rej_f   = (const int*)d_in[29];   const int* rej_i   = (const int*)d_in[30];

    int E_INT = in_sizes[19], E_FRD = in_sizes[21], E_LIK = in_sizes[23], E_BEL = in_sizes[25];

    // ---- workspace carving (floats)
    float* ws = (float*)d_ws;
    float* deg_frd_out = ws;                      // U
    float* deg_frd_in  = deg_frd_out + U_;        // U
    float* deg_int_out = deg_frd_in  + U_;        // U
    float* deg_int_in  = deg_int_out + U_;        // NI
    float* deg_bel_out = deg_int_in  + NI_;       // NA
    float* deg_bel_in  = deg_bel_out + NA_;       // NI
    float* deg_lik_out = deg_bel_in  + NI_;       // U
    float* deg_lik_in  = deg_lik_out + U_;        // NA
    float* agg_frd = deg_lik_in + NA_;            // U  x 64
    float* agg_int = agg_frd + (size_t)U_  * EMB; // NI x 64
    float* agg_bel = agg_int + (size_t)NI_ * EMB; // NI x 64
    float* agg_lik = agg_bel + (size_t)NI_ * EMB; // NA x 64
    float* gnn     = agg_lik + (size_t)NA_ * EMB; // 170000 x 100
    float* support = gnn + (size_t)(U_ + NI_ + NA_) * HID;  // 8192 x 100
    float* interest = support + (size_t)B_ * N_ * HID;      // 32 x 257 x 100

    // zero degree + agg region
    size_t zero_floats = (size_t)(4 * U_ + 2 * NI_ + 2 * NA_) +
                         (size_t)(U_ + 2 * NI_ + NA_) * EMB;
    hipMemsetAsync(ws, 0, zero_floats * sizeof(float), stream);

    // ---- degrees
    auto cdiv = [](long long a, long long b) { return (int)((a + b - 1) / b); };
    count_deg_kernel<<<cdiv(E_FRD, 256), 256, 0, stream>>>(frd_src, E_FRD, deg_frd_out);
    count_deg_kernel<<<cdiv(E_FRD, 256), 256, 0, stream>>>(frd_dst, E_FRD, deg_frd_in);
    count_deg_kernel<<<cdiv(E_INT, 256), 256, 0, stream>>>(int_src, E_INT, deg_int_out);
    count_deg_kernel<<<cdiv(E_INT, 256), 256, 0, stream>>>(int_dst, E_INT, deg_int_in);
    count_deg_kernel<<<cdiv(E_BEL, 256), 256, 0, stream>>>(bel_src, E_BEL, deg_bel_out);
    count_deg_kernel<<<cdiv(E_BEL, 256), 256, 0, stream>>>(bel_dst, E_BEL, deg_bel_in);
    count_deg_kernel<<<cdiv(E_LIK, 256), 256, 0, stream>>>(lik_src, E_LIK, deg_lik_out);
    count_deg_kernel<<<cdiv(E_LIK, 256), 256, 0, stream>>>(lik_dst, E_LIK, deg_lik_in);

    // ---- edge aggregation (h0_user = emb rows [0,U); h0_attr = rows [U+NI, ...))
    const float* h0_user = emb;
    const float* h0_attr = emb + (size_t)(U_ + NI_) * EMB;
    edge_aggregate_kernel<<<cdiv((long long)E_FRD * (EMB / 4), 256), 256, 0, stream>>>(
        frd_src, frd_dst, E_FRD, h0_user, deg_frd_out, agg_frd);
    edge_aggregate_kernel<<<cdiv((long long)E_INT * (EMB / 4), 256), 256, 0, stream>>>(
        int_src, int_dst, E_INT, h0_user, deg_int_out, agg_int);
    edge_aggregate_kernel<<<cdiv((long long)E_BEL * (EMB / 4), 256), 256, 0, stream>>>(
        bel_src, bel_dst, E_BEL, h0_attr, deg_bel_out, agg_bel);
    edge_aggregate_kernel<<<cdiv((long long)E_LIK * (EMB / 4), 256), 256, 0, stream>>>(
        lik_src, lik_dst, E_LIK, h0_user, deg_lik_out, agg_lik);

    // ---- graphconv projections (WMMA), fused scale/bias/lrelu/combine
    dim3 blk(128);
    auto grid_m = [&](int M) { return dim3((unsigned)((M / 16 + 3) / 4), 7); };
    gemm_rows_wmma<0, false, true, true><<<grid_m(U_), blk, 0, stream>>>(
        agg_frd, nullptr, deg_frd_in, W_frd, b_frd, gnn, U_);
    gemm_rows_wmma<1, false, true, true><<<grid_m(NI_), blk, 0, stream>>>(
        agg_int, nullptr, deg_int_in, W_int, b_int, gnn + (size_t)U_ * HID, NI_);
    gemm_rows_wmma<2, false, true, true><<<grid_m(NI_), blk, 0, stream>>>(
        agg_bel, nullptr, deg_bel_in, W_bel, b_bel, gnn + (size_t)U_ * HID, NI_);
    gemm_rows_wmma<0, false, true, true><<<grid_m(NA_), blk, 0, stream>>>(
        agg_lik, nullptr, deg_lik_in, W_lik, b_lik, gnn + (size_t)(U_ + NI_) * HID, NA_);

    // ---- support = emb[neighbors] @ gcn_W (WMMA, gathered rows, no bias)
    gemm_rows_wmma<3, true, false, false><<<grid_m(B_ * N_), blk, 0, stream>>>(
        emb, neighbors, nullptr, gcn_W, nullptr, support, B_ * N_);

    // ---- seq = (1-eps)*(adj@support + gcn_b) + eps*gnn[cur_node] -> interest[:, 0:256, :]
    gemm_adj_wmma<<<dim3(B_ * 7), blk, 0, stream>>>(adj, support, gcn_b, eps_p,
                                                    gnn, cur_node, interest);

    // ---- mm row -> interest[:, 256, :]
    rej_mm_kernel<<<B_, 128, 0, stream>>>(gnn, rej_f, rej_i, fcW, fcb, interest);

    // ---- two-pass attention -> d_out (4, 32, 1, 100)
    attention_kernel<<<4 * B_, 512, 0, stream>>>(interest, aiW, aib, aaW, aab, (float*)d_out);
}
